// UniformRegularizationLoss_20753281974357
// MI455X (gfx1250) — compile-verified, matrix operand order checked by reading, not run on hardware
//
#include <hip/hip_runtime.h>
#include <hip/hip_bf16.h>
#include <hip/hip_fp16.h>

typedef __attribute__((ext_vector_type(16))) _Float16 v16h;
typedef __attribute__((ext_vector_type(8)))  _Float16 v8h;
typedef __attribute__((ext_vector_type(8)))  float    v8f;
typedef __attribute__((ext_vector_type(4)))  float    v4f;

#define MATN    8192
#define KSLICES 8
#define SLICE   (MATN / KSLICES)       // 1024 columns per slice
#define BETA_C  0.3f

// ---------------------------------------------------------------- init u,v = 0
__global__ void k_init(float* __restrict__ u, float* __restrict__ v) {
  int i = blockIdx.x * blockDim.x + threadIdx.x;
  if (i < MATN) { u[i] = 0.0f; v[i] = 0.0f; }
}

// ------------------------------------------------- f32 -> f16 convert (stream)
__global__ __launch_bounds__(256)
void k_convert(const float* __restrict__ S, _Float16* __restrict__ Sh) {
  long long i = ((long long)blockIdx.x * blockDim.x + threadIdx.x) * 8ll;
  v4f a = *(const v4f*)(S + i);
  v4f b = *(const v4f*)(S + i + 4);
  v8h o;
  o[0] = (_Float16)a[0]; o[1] = (_Float16)a[1];
  o[2] = (_Float16)a[2]; o[3] = (_Float16)a[3];
  o[4] = (_Float16)b[0]; o[5] = (_Float16)b[1];
  o[6] = (_Float16)b[2]; o[7] = (_Float16)b[3];
  *(v8h*)(Sh + i) = o;
}

// ------------------------------------- f32 -> f16 transpose via LDS 32x32 tile
__global__ __launch_bounds__(256)
void k_transpose(const float* __restrict__ S, _Float16* __restrict__ ShT) {
  __shared__ float tile[32][33];
  int x = threadIdx.x;          // 0..31
  int y = threadIdx.y;          // 0..7
  int col  = blockIdx.x * 32 + x;
  int row0 = blockIdx.y * 32;
  #pragma unroll
  for (int j = 0; j < 32; j += 8)
    tile[y + j][x] = S[(size_t)(row0 + y + j) * MATN + col];
  __syncthreads();
  int ocol  = blockIdx.y * 32 + x;   // transposed column = original row
  int orow0 = blockIdx.x * 32;       // transposed row    = original column
  #pragma unroll
  for (int j = 0; j < 32; j += 8)
    ShT[(size_t)(orow0 + y + j) * MATN + ocol] = (_Float16)tile[x][y + j];
}

// ----------------------------------------- max(vec) -> scalar (single block)
__global__ __launch_bounds__(256)
void k_vecmax(const float* __restrict__ vec, float* __restrict__ out) {
  float m = -3.0e38f;
  for (int i = threadIdx.x; i < MATN; i += 256) m = fmaxf(m, vec[i]);
  #pragma unroll
  for (int o = 16; o > 0; o >>= 1) m = fmaxf(m, __shfl_xor(m, o, 32));
  __shared__ float sm[8];
  if ((threadIdx.x & 31) == 0) sm[threadIdx.x >> 5] = m;
  __syncthreads();
  if (threadIdx.x == 0) {
    float M = sm[0];
    #pragma unroll
    for (int i = 1; i < 8; ++i) M = fmaxf(M, sm[i]);
    out[0] = M;
  }
}

// ---------------------------------------------------------------------------
// Sinkhorn half-step, K-sliced for chip saturation:
//   partial[slice][row] = sum_{j in slice} exp(P[row][j] + vin[j] - shift)
// Each wave owns a 16-row strip over one 1024-column slice and reduces with
// V_WMMA_F32_16X16X32_F16 against a ones B-matrix (D[m][n] = row-sum m).
// A-layout per ISA: lanes 0-15 -> M=lane, K in {0..7,16..23};
//                   lanes 16-31 -> M=lane-16, K in {8..15,24..31}.
// shift = max(vin) keeps exp() residuals inside f16 range (v-update feeds
// u ~ -19 which would otherwise flush to zero in f16).
// ---------------------------------------------------------------------------
__global__ __launch_bounds__(128)
void k_sinkhorn_partial(const _Float16* __restrict__ P,
                        const float* __restrict__ vin,
                        const float* __restrict__ shift_p,
                        float* __restrict__ partial) {
  __shared__ float ldsv[SLICE];                      // 4 KB slice of dual vector
  const int slice   = blockIdx.y;
  const int colBase = slice * SLICE;
  const float shift = *shift_p;
  for (int i = threadIdx.x; i < SLICE; i += 128)
    ldsv[i] = vin[colBase + i] - shift;
  __syncthreads();

  const int lane    = threadIdx.x & 31;
  const int wave    = threadIdx.x >> 5;
  const int rowBase = (blockIdx.x * 4 + wave) * 16;
  const int mrow    = lane & 15;
  const int off     = (lane >> 4) << 3;              // 0 or 8
  const _Float16* rp = P + (size_t)(rowBase + mrow) * MATN + colBase;

  v8f  c = {};
  v16h ones;
  #pragma unroll
  for (int i = 0; i < 16; ++i) ones[i] = (_Float16)1.0f;

  for (int k = 0; k < SLICE; k += 32) {
    __builtin_prefetch((const void*)(rp + k + 512), 0, 0); // global_prefetch_b8
    v8h a0 = *(const v8h*)(rp + k + off);             // K = off .. off+7
    v8h a1 = *(const v8h*)(rp + k + 16 + off);        // K = 16+off .. 16+off+7
    v4f v00 = *(const v4f*)&ldsv[k + off];
    v4f v01 = *(const v4f*)&ldsv[k + off + 4];
    v4f v10 = *(const v4f*)&ldsv[k + 16 + off];
    v4f v11 = *(const v4f*)&ldsv[k + 16 + off + 4];
    v16h A;
    #pragma unroll
    for (int i = 0; i < 4; ++i) {
      A[i]      = (_Float16)__expf((float)a0[i]     + v00[i]);
      A[i + 4]  = (_Float16)__expf((float)a0[i + 4] + v01[i]);
      A[i + 8]  = (_Float16)__expf((float)a1[i]     + v10[i]);
      A[i + 12] = (_Float16)__expf((float)a1[i + 4] + v11[i]);
    }
    c = __builtin_amdgcn_wmma_f32_16x16x32_f16(
        /*neg_a=*/false, A, /*neg_b=*/false, ones,
        /*c_mod=*/(short)0, c, /*reuse_a=*/false, /*reuse_b=*/false);
  }

  // C layout: VGPR r holds row M=r (lanes 0-15) / M=8+r (lanes 16-31),
  // replicated across the 16 N-lanes. Lane 0 emits rows 0-7, lane 16 rows 8-15.
  if ((lane & 15) == 0) {
    int base = rowBase + ((lane >> 4) << 3);
    #pragma unroll
    for (int r = 0; r < 8; ++r)
      partial[(size_t)slice * MATN + base + r] = c[r];
  }
}

// -------------------- combine slice partials: u = norm - shift - log(sum)
__global__ __launch_bounds__(256)
void k_combine(const float* __restrict__ partial, float* __restrict__ uout,
               const float* __restrict__ shift_p, float norm) {
  int i = blockIdx.x * blockDim.x + threadIdx.x;
  if (i < MATN) {
    float s = 0.0f;
    #pragma unroll
    for (int k = 0; k < KSLICES; ++k) s += partial[(size_t)k * MATN + i];
    uout[i] = norm - shift_p[0] - __logf(s);
  }
}

// --------------------------------- per-row LSE of scale*S with online max-shift
__global__ __launch_bounds__(256)
void k_lse_rows(const float* __restrict__ S, const float* __restrict__ scale_p,
                float* __restrict__ Lrow) {
  const int row = blockIdx.x;
  const float scale = *scale_p;
  const float* rp = S + (size_t)row * MATN;
  float m = -3.0e38f, s = 0.0f;
  for (int j = threadIdx.x; j < MATN; j += 256) {
    float x = scale * rp[j];
    if (x > m) { s = s * __expf(m - x) + 1.0f; m = x; }
    else        s += __expf(x - m);
  }
  #pragma unroll
  for (int o = 16; o > 0; o >>= 1) {
    float m2 = __shfl_xor(m, o, 32);
    float s2 = __shfl_xor(s, o, 32);
    float M  = fmaxf(m, m2);
    s = s * __expf(m - M) + s2 * __expf(m2 - M);
    m = M;
  }
  __shared__ float sm[8], ss[8];
  if ((threadIdx.x & 31) == 0) { sm[threadIdx.x >> 5] = m; ss[threadIdx.x >> 5] = s; }
  __syncthreads();
  if (threadIdx.x == 0) {
    float M = sm[0];
    #pragma unroll
    for (int i = 1; i < 8; ++i) M = fmaxf(M, sm[i]);
    float acc = 0.0f;
    #pragma unroll
    for (int i = 0; i < 8; ++i) acc += ss[i] * __expf(sm[i] - M);
    Lrow[row] = M + __logf(acc);
  }
}

// ------------- per-row loss: -( beta*sum_j (s*S-L)*Q + (1-beta)*(s*S_ii - L) )
__global__ __launch_bounds__(256)
void k_row_loss(const float* __restrict__ S, const float* __restrict__ u,
                const float* __restrict__ v, const float* __restrict__ Lrow,
                const float* __restrict__ scale_p, float* __restrict__ per_row,
                float norm) {
  const int row = blockIdx.x;
  const float scale = *scale_p;
  const float* rp = S + (size_t)row * MATN;
  const float L = Lrow[row];
  const float ui = u[row];
  float acc = 0.0f;
  for (int j = threadIdx.x; j < MATN; j += 256) {
    float sij = rp[j];
    float q = __expf(sij + ui + v[j] - norm);   // transport plan Q_ij
    acc += (scale * sij - L) * q;               // log_prob * Q
  }
  #pragma unroll
  for (int o = 16; o > 0; o >>= 1) acc += __shfl_xor(acc, o, 32);
  __shared__ float sa[8];
  if ((threadIdx.x & 31) == 0) sa[threadIdx.x >> 5] = acc;
  __syncthreads();
  if (threadIdx.x == 0) {
    float tot = 0.0f;
    #pragma unroll
    for (int i = 0; i < 8; ++i) tot += sa[i];
    float diag = scale * rp[row] - L;           // (1-beta)*eye term
    per_row[row] = -(BETA_C * tot + (1.0f - BETA_C) * diag);
  }
}

// --------------------------------------------- deterministic final mean-reduce
__global__ __launch_bounds__(256)
void k_final(const float* __restrict__ per_row, float* __restrict__ out) {
  __shared__ float sa[256];
  float a = 0.0f;
  for (int i = threadIdx.x; i < MATN; i += 256) a += per_row[i];
  sa[threadIdx.x] = a;
  __syncthreads();
  for (int s = 128; s > 0; s >>= 1) {
    if (threadIdx.x < s) sa[threadIdx.x] += sa[threadIdx.x + s];
    __syncthreads();
  }
  if (threadIdx.x == 0) out[0] = sa[0] / (float)MATN;
}

extern "C" void kernel_launch(void* const* d_in, const int* in_sizes, int n_in,
                              void* d_out, int out_size, void* d_ws, size_t ws_size,
                              hipStream_t stream) {
  const float* S       = (const float*)d_in[0];   // 8192*8192 f32
  const float* scale_p = (const float*)d_in[1];   // scalar logit_scale
  float* out = (float*)d_out;

  // Workspace: f16 copy (128MB) | f16 transpose (128MB) | partial | u,v,L,pr,shift
  char* ws = (char*)d_ws;
  const size_t matElems = (size_t)MATN * MATN;
  _Float16* Sh  = (_Float16*)ws;
  _Float16* ShT = (_Float16*)(ws + matElems * 2);
  float* partial = (float*)(ws + matElems * 4);   // KSLICES * MATN floats (256KB)
  float* u     = partial + (size_t)KSLICES * MATN;
  float* v     = u + MATN;
  float* Lr    = v + MATN;
  float* pr    = Lr + MATN;
  float* shift = pr + MATN;                       // 1 float

  const float norm = -logf((float)(MATN + MATN)); // log(1/(m+n))

  k_init<<<MATN / 256, 256, 0, stream>>>(u, v);
  k_convert<<<(int)(matElems / 8 / 256), 256, 0, stream>>>(S, Sh);
  k_transpose<<<dim3(MATN / 32, MATN / 32), dim3(32, 8), 0, stream>>>(S, ShT);

  const dim3 passGrid(MATN / (16 * 4), KSLICES); // 16 rows/wave * 4 waves, 8 K-slices
  for (int it = 0; it < 50; ++it) {
    // u update: row LSE of (S + v)
    k_vecmax<<<1, 256, 0, stream>>>(v, shift);
    k_sinkhorn_partial<<<passGrid, 128, 0, stream>>>(Sh, v, shift, partial);
    k_combine<<<MATN / 256, 256, 0, stream>>>(partial, u, shift, norm);
    // v update: column LSE of (S + u) == row LSE over transposed copy
    k_vecmax<<<1, 256, 0, stream>>>(u, shift);
    k_sinkhorn_partial<<<passGrid, 128, 0, stream>>>(ShT, u, shift, partial);
    k_combine<<<MATN / 256, 256, 0, stream>>>(partial, v, shift, norm);
  }

  k_lse_rows<<<MATN, 256, 0, stream>>>(S, scale_p, Lr);
  k_row_loss<<<MATN, 256, 0, stream>>>(S, u, v, Lr, scale_p, pr, norm);
  k_final<<<1, 256, 0, stream>>>(pr, out);
}
